// OcclusionDetector_6141803233764
// MI455X (gfx1250) — matrix operand order repaired and checked by reading, not compiled, not run on hardware
//
#include <hip/hip_runtime.h>
#include <hip/hip_bf16.h>
#include <stdint.h>
#include <math.h>

// ---------------- problem constants (from the reference) ----------------
#define S_INIT   64          // init_samples
#define NR       2048        // Np*Nc = 512*4 rays
#define HID      128
#define MAXNS    256         // Ns after the last upsample
#define MIN_DIST 0.1f
#define EPS_ERR  0.1f

typedef __attribute__((ext_vector_type(16))) __bf16 v16bf;
typedef __attribute__((ext_vector_type(8)))  float  v8f;

#if defined(__HIP_DEVICE_COMPILE__) && __has_builtin(__builtin_amdgcn_tensor_load_to_lds) && \
    __has_builtin(__builtin_amdgcn_s_wait_tensorcnt)
#define USE_TDM 1
typedef uint32_t u32x4 __attribute__((ext_vector_type(4)));
typedef int32_t  i32x4 __attribute__((ext_vector_type(4)));
typedef int32_t  i32x8 __attribute__((ext_vector_type(8)));
#else
#define USE_TDM 0
#endif

// ---------------- device helpers ----------------
__device__ __forceinline__ float softplusf(float x) {
  // numerically stable log1p(exp(x)) = max(x,0) + log1p(exp(-|x|))
  return fmaxf(x, 0.f) + log1pf(expf(-fabsf(x)));
}
__device__ __forceinline__ float vol_sigma(float s, float beta) {
  float psi = (s >= 0.f) ? 0.5f * expf(-s / beta) : 1.f - 0.5f * expf(s / beta);
  return psi / beta;
}
__device__ __forceinline__ float d_star_f(float a, float s0, float s1) {
  float b = fabsf(s0), c = fabsf(s1);
  float ds;
  if (a * a + b * b <= c * c)      ds = b;
  else if (a * a + c * c <= b * b) ds = c;
  else if (b + c - a > 0.f) {
    float s  = 0.5f * (a + b + c);
    float ar = fmaxf(s * (s - a) * (s - b) * (s - c), 0.f);
    ds = 2.f * sqrtf(ar) / fmaxf(a, 1e-10f);
  } else ds = 0.f;
  bool same_sign = (s0 > 0.f && s1 > 0.f) || (s0 < 0.f && s1 < 0.f);
  return same_sign ? ds : 0.f;
}

// ---------------- setup kernels ----------------
__global__ void k_init(const float* __restrict__ cam, const float* __restrict__ ipts,
                       const unsigned char* __restrict__ insrc,
                       float* __restrict__ dirs, float* __restrict__ tdist,
                       float* __restrict__ sdf, int* __restrict__ mask) {
  int r = blockIdx.x * blockDim.x + threadIdx.x;
  if (r >= NR) return;
  int p = r >> 2, c = r & 3;
  float rx = ipts[3 * p + 0] - cam[3 * c + 0];
  float ry = ipts[3 * p + 1] - cam[3 * c + 1];
  float rz = ipts[3 * p + 2] - cam[3 * c + 2];
  float nrm = sqrtf(rx * rx + ry * ry + rz * rz);
  float inv = 1.f / fmaxf(nrm, 1e-12f);
  dirs[3 * r + 0] = rx * inv; dirs[3 * r + 1] = ry * inv; dirs[3 * r + 2] = rz * inv;
  float md = nrm - MIN_DIST;
  int ok = insrc[r] ? 1 : 0;
  for (int s = 0; s < S_INIT; ++s) {
    tdist[r * S_INIT + s] = ((float)s / (float)(S_INIT - 1)) * md;
    sdf[r * S_INIT + s]   = 0.f;
    mask[r * S_INIT + s]  = ok;
  }
}

__global__ void k_convert(const float* __restrict__ W1, const float* __restrict__ W2,
                          __bf16* __restrict__ W1t, __bf16* __restrict__ W2t) {
  // transpose (k,n) -> (n,k) so WMMA B-fragment K-pairs are contiguous in LDS
  int g = blockIdx.x * blockDim.x + threadIdx.x;
  if (g < HID * HID) {
    int n = g / HID, k = g % HID;
    W1t[n * HID + k] = (__bf16)W1[k * HID + n];
  } else if (g < 2 * HID * HID) {
    int q = g - HID * HID;
    int n = q / HID, k = q % HID;
    W2t[n * HID + k] = (__bf16)W2[k * HID + n];
  }
}

__global__ void k_gen_points(const float* __restrict__ cam, const float* __restrict__ dirs,
                             const float* __restrict__ tdist, float* __restrict__ pts, int Ns) {
  int g = blockIdx.x * blockDim.x + threadIdx.x;
  if (g >= NR * Ns) return;
  int r = g / Ns;
  int c = r & 3;
  float t = tdist[g];
  pts[3 * g + 0] = cam[3 * c + 0] + t * dirs[3 * r + 0];
  pts[3 * g + 1] = cam[3 * c + 1] + t * dirs[3 * r + 1];
  pts[3 * g + 2] = cam[3 * c + 2] + t * dirs[3 * r + 2];
}

// ---------------- fused WMMA MLP: sdf_fn over a tile of points ----------------
// 4 waves/block, 32 points/wave (two 16-row M-tiles share every B fragment),
// weights double-buffered in LDS and fetched by the Tensor Data Mover.
#define MLP_WAVES 4
#define MLP_TPB   (MLP_WAVES * 32)
#define MLP_PTS   (MLP_WAVES * 32)

__global__ __launch_bounds__(MLP_TPB) void k_mlp(
    const float* __restrict__ pts, const int* __restrict__ mask,
    float* __restrict__ sdf,
    const float* __restrict__ W0, const float* __restrict__ b0,
    const __bf16* __restrict__ W1t, const float* __restrict__ b1,
    const __bf16* __restrict__ W2t, const float* __restrict__ b2,
    const float* __restrict__ W3, const float* __restrict__ b3) {
  __shared__ __bf16 sW[2][HID][HID];              // 64 KB: both layers' weights, [n][k]
  __shared__ __bf16 sAct[MLP_WAVES][32][HID];     // 32 KB: wave-private activations
  __shared__ float  sW0c[3 * HID];
  __shared__ float  sB0[HID];
  __shared__ float  sB1[HID];
  __shared__ float  sB2[HID];
  __shared__ float  sW3[HID];

  const int tid  = threadIdx.x;
  const int lane = tid & 31;
  const int wv   = tid >> 5;
  const int hh   = lane >> 4;      // lane half (ISA A/C layout split)
  const int m    = lane & 15;

  // stage small constants (all layers' biases up front: no mid-kernel reloads)
  for (int i = tid; i < 3 * HID; i += MLP_TPB) sW0c[i] = W0[i];
  for (int i = tid; i < HID; i += MLP_TPB) {
    sB0[i] = b0[i]; sB1[i] = b1[i]; sB2[i] = b2[i]; sW3[i] = W3[i];
  }

#if USE_TDM
  // ---- Tensor Data Mover: DMA both 32KB weight tiles, overlap with layer 0 ----
  if (tid < 32) {                      // wave 0 issues; TENSORcnt is per-wave, in-order
    // D# group1: 1-row tile of 8192 dwords (data_size=4B), no multicast/pad/iterate
    i32x8 g1;
    g1[0] = 0x00020000;                // data_size = 2 (4 bytes)
    g1[1] = 0x20000000;                // tensor_dim0 = 8192 (bits 79:48, low half)
    g1[2] = 0x00010000;                // tensor_dim0 hi = 0, tensor_dim1 = 1
    g1[3] = 0x20000000;                // tile_dim0 = 8192
    g1[4] = 0x00000001;                // tile_dim1 = 1, tile_dim2 = 0
    g1[5] = 8192;                      // tensor_dim0_stride lo
    g1[6] = 0x20000000;                // tensor_dim0_stride hi = 0, dim1_stride lo = 8192
    g1[7] = 0;
    i32x4 z4; z4[0] = z4[1] = z4[2] = z4[3] = 0;
    i32x8 z8;
#pragma unroll
    for (int i = 0; i < 8; ++i) z8[i] = 0;
    uint64_t ga0 = (uint64_t)(uintptr_t)W1t;
    uint64_t ga1 = (uint64_t)(uintptr_t)W2t;
    uint32_t ld0 = (uint32_t)(uintptr_t)&sW[0][0][0];   // LDS byte offset
    uint32_t ld1 = (uint32_t)(uintptr_t)&sW[1][0][0];
    u32x4 g0a, g0b;
    g0a[0] = 1u;  g0a[1] = ld0; g0a[2] = (uint32_t)ga0;
    g0a[3] = (uint32_t)(ga0 >> 32) | (2u << 30);        // type = 2 ("image")
    g0b[0] = 1u;  g0b[1] = ld1; g0b[2] = (uint32_t)ga1;
    g0b[3] = (uint32_t)(ga1 >> 32) | (2u << 30);
    __builtin_amdgcn_tensor_load_to_lds(g0a, g1, z4, z4, z8, 0);
    __builtin_amdgcn_tensor_load_to_lds(g0b, g1, z4, z4, z8, 0);
  }
#else
  {
    const uint32_t* s1 = (const uint32_t*)W1t;
    const uint32_t* s2 = (const uint32_t*)W2t;
    uint32_t* dst = (uint32_t*)&sW[0][0][0];
    for (int i = tid; i < HID * HID / 2; i += MLP_TPB) {
      dst[i] = s1[i];
      dst[HID * HID / 2 + i] = s2[i];
    }
  }
#endif
  __syncthreads();                     // constants visible to all waves

  // ---- layer 0: 3 -> 128 (K too small for WMMA; VALU, overlaps the weight DMA) ----
  const int pbase = blockIdx.x * MLP_PTS + wv * 32;    // grid is an exact multiple
#pragma unroll
  for (int T = 0; T < 2; ++T) {
    const int p = pbase + T * 16 + m;
    const float x = pts[3 * p + 0], y = pts[3 * p + 1], z = pts[3 * p + 2];
#pragma unroll 4
    for (int i = 0; i < 64; ++i) {
      int n = hh * 64 + i;
      float h = sW0c[n] * x + sW0c[HID + n] * y + sW0c[2 * HID + n] * z + sB0[n];
      sAct[wv][T * 16 + m][n] = (__bf16)softplusf(h);
    }
  }

#if USE_TDM
  if (tid < 32) __builtin_amdgcn_s_wait_tensorcnt(1);  // first tile (layer-1 W) done
#endif
  __syncthreads();

  // ---- layers 1,2: 128 -> 128 via v_wmma_f32_16x16x32_bf16 ----
#pragma unroll 1
  for (int L = 0; L < 2; ++L) {
    const float* bb = (L == 0) ? sB1 : sB2;
    v8f acc[2][8];
#pragma unroll
    for (int t = 0; t < 8; ++t) {
      float bv = bb[t * 16 + m];         // C layout: col N = lane&15
#pragma unroll
      for (int rr = 0; rr < 8; ++rr) { acc[0][t][rr] = bv; acc[1][t][rr] = bv; }
    }
#pragma unroll
    for (int kk = 0; kk < 4; ++kk) {
      // A fragments (16x32 bf16, ISA layout): lane<16 holds K 0-7/16-23, lane>=16 K 8-15/24-31
      v16bf aF[2];
#pragma unroll
      for (int T = 0; T < 2; ++T) {
#pragma unroll
        for (int v = 0; v < 8; ++v) {
          int kb = kk * 32 + ((v & 3) * 2) + ((v >= 4) ? 16 : 0) + hh * 8;
          aF[T][2 * v]     = sAct[wv][T * 16 + m][kb];
          aF[T][2 * v + 1] = sAct[wv][T * 16 + m][kb + 1];
        }
      }
#pragma unroll
      for (int t = 0; t < 8; ++t) {
        // B fragment (32x16 bf16): per VGPR a (K,K+1) pair for col N = lane&15,
        // shared by both M-tiles (halves LDS B-traffic per WMMA)
        v16bf bF;
#pragma unroll
        for (int v = 0; v < 8; ++v) {
          int kb = kk * 32 + ((v & 3) * 2) + ((v >= 4) ? 16 : 0) + hh * 8;
          bF[2 * v]     = sW[L][t * 16 + m][kb];
          bF[2 * v + 1] = sW[L][t * 16 + m][kb + 1];
        }
        acc[0][t] = __builtin_amdgcn_wmma_f32_16x16x32_bf16(
            false, aF[0], false, bF, (short)0, acc[0][t], false, false);
        acc[1][t] = __builtin_amdgcn_wmma_f32_16x16x32_bf16(
            false, aF[1], false, bF, (short)0, acc[1][t], false, false);
      }
    }
    // softplus + write next-layer activations (wave-private region; per-wave DS order)
#pragma unroll
    for (int T = 0; T < 2; ++T) {
#pragma unroll
      for (int t = 0; t < 8; ++t) {
#pragma unroll
        for (int rr = 0; rr < 8; ++rr) {
          int mr = T * 16 + rr + 8 * hh;   // C layout: row M = r + 8*half
          sAct[wv][mr][t * 16 + m] = (__bf16)softplusf(acc[T][t][rr]);
        }
      }
    }
    if (L == 0) {
#if USE_TDM
      if (tid < 32) __builtin_amdgcn_s_wait_tensorcnt(0);  // layer-2 weights done
#endif
      __syncthreads();
    }
  }

  // ---- layer 3: 128 -> 1 (VALU dot; all 32 lanes busy, one point each) ----
  {
    float a = b3[0];
#pragma unroll 8
    for (int k = 0; k < HID; ++k) a += (float)sAct[wv][lane][k] * sW3[k];
    int p = pbase + lane;
    if (mask[p]) sdf[p] = a;             // masked re-eval (dense + where)
  }
}

// ---------------- per-ray serial kernels (tiny fraction of runtime) ----------------
__global__ void k_error_ok(const float* __restrict__ sdf, const float* __restrict__ tdist,
                           const float* __restrict__ betap, int Ns, int* __restrict__ eok) {
  int r = blockIdx.x * blockDim.x + threadIdx.x;
  if (r >= NR) return;
  float beta = betap[0];
  const float* sd = sdf + r * Ns;
  const float* td = tdist + r * Ns;
  float errint = 0.f, fesum = 0.f, last = 0.f;
  for (int j = 0; j < Ns - 1; ++j) {
    float dl  = td[j + 1] - td[j];
    float ds  = d_star_f(dl, sd[j], sd[j + 1]);
    errint   += expf(-ds / beta) * dl * dl / (4.f * beta * beta);
    last      = (fminf(expf(errint), 1e6f) - 1.f) * expf(-fesum);
    fesum    += vol_sigma(sd[j], beta) * dl;
  }
  eok[r] = (last < EPS_ERR) ? 1 : 0;
}

__global__ void k_upsample(const float* __restrict__ sdf_in, const float* __restrict__ t_in,
                           const int* __restrict__ eokp, const unsigned char* __restrict__ insrc,
                           const float* __restrict__ betap, int Ns_in,
                           float* __restrict__ sdf_out, float* __restrict__ t_out,
                           int* __restrict__ mask_out, int* __restrict__ nb_buf,
                           float* __restrict__ frac_buf) {
  int r = blockIdx.x * blockDim.x + threadIdx.x;
  if (r >= NR) return;
  float beta = betap[0];
  const float* sd = sdf_in + r * Ns_in;
  const float* td = t_in + r * Ns_in;
  int*   nb = nb_buf + r * MAXNS;
  float* fr = frac_buf + r * MAXNS;
  const int NI = Ns_in - 1;
  // cumulative VolSDF interval error (clipped at 100) and its sum
  float errint = 0.f, fesum = 0.f, tot = 0.f;
  for (int j = 0; j < NI; ++j) {
    float dl = td[j + 1] - td[j];
    float ds = d_star_f(dl, sd[j], sd[j + 1]);
    errint  += expf(-ds / beta) * dl * dl / (4.f * beta * beta);
    float ie = fminf((fminf(expf(errint), 1e6f) - 1.f) * expf(-fesum), 100.f);
    fr[j] = ie; tot += ie;
    fesum += vol_sigma(sd[j], beta) * dl;
  }
  float inv = (float)S_INIT / (tot + 1e-6f);
  int sumnb = 0;
  for (int j = 0; j < NI; ++j) {
    float ep = fr[j] * inv;
    float fl = floorf(ep);
    nb[j] = (int)fl;
    fr[j] = ep - fl;
    sumnb += nb[j];
  }
  // top-k on fractional parts: +1 to the `remaining` largest (ties -> lowest index)
  int rem = S_INIT - sumnb;
  int K   = (NI < S_INIT) ? NI : S_INIT;
  int sel = rem < K ? rem : K; if (sel < 0) sel = 0;
  for (int s = 0; s < sel; ++s) {
    int arg = 0; float best = -1.f;
    for (int j = 0; j < NI; ++j) if (fr[j] > best) { best = fr[j]; arg = j; }
    fr[arg] = -2.f; nb[arg] += 1; sumnb += 1;
  }
  nb[0] += S_INIT - sumnb;                 // force sum == S
  // emit new samples: per interval offsets 0..nb[j], then the closing endpoint
  int eok = eokp[r];
  int ok  = insrc[r] ? 1 : 0;
  int Ns_out = Ns_in + S_INIT;
  float* so = sdf_out + r * Ns_out;
  float* to = t_out + r * Ns_out;
  int*   mo = mask_out + r * Ns_out;
  int o = 0;
  for (int j = 0; j < NI; ++j) {
    float dl    = td[j + 1] - td[j];
    float denom = (float)(nb[j] + 1);
    float tstep = dl / denom;
    float sstep = (sd[j + 1] - sd[j]) / denom;
    for (int q = 0; q <= nb[j]; ++q) {
      to[o] = td[j] + (float)q * tstep;
      so[o] = sd[j] + (float)q * sstep;
      mo[o] = (q > 0 && !eok && ok) ? 1 : 0;
      ++o;
    }
  }
  to[o] = td[NI]; so[o] = sd[NI]; mo[o] = 0;
}

__global__ void k_final(const float* __restrict__ sdf, const float* __restrict__ tdist,
                        const unsigned char* __restrict__ insrc,
                        const float* __restrict__ betap, int Ns, float* __restrict__ out) {
  int r = blockIdx.x * blockDim.x + threadIdx.x;
  if (r >= NR) return;
  float beta = betap[0];
  const float* sd = sdf + r * Ns;
  const float* td = tdist + r * Ns;
  float acc = 0.f;
  for (int j = 0; j < Ns - 1; ++j) acc += vol_sigma(sd[j], beta) * (td[j + 1] - td[j]);
  // 1 - prod(exp(-sigma*delta)) == 1 - exp(-sum)
  out[r] = insrc[r] ? (1.f - expf(-acc)) : 0.f;
}

// ---------------- host orchestration ----------------
extern "C" void kernel_launch(void* const* d_in, const int* in_sizes, int n_in,
                              void* d_out, int out_size, void* d_ws, size_t ws_size,
                              hipStream_t stream) {
  (void)in_sizes; (void)n_in; (void)out_size; (void)ws_size;
  const float* cam  = (const float*)d_in[0];
  const float* ipts = (const float*)d_in[1];
  const unsigned char* insrc = (const unsigned char*)d_in[2];   // numpy bool = 1 byte
  const float* W0 = (const float*)d_in[3];  const float* b0 = (const float*)d_in[4];
  const float* W1 = (const float*)d_in[5];  const float* b1 = (const float*)d_in[6];
  const float* W2 = (const float*)d_in[7];  const float* b2 = (const float*)d_in[8];
  const float* W3 = (const float*)d_in[9];  const float* b3 = (const float*)d_in[10];
  const float* beta = (const float*)d_in[11];

  char* ws = (char*)d_ws;
  size_t off = 0;
  auto carve = [&](size_t bytes) -> char* {
    char* p = ws + off;
    off = (off + bytes + 255) & ~(size_t)255;
    return p;
  };
  float* dirs  = (float*)carve((size_t)NR * 3 * 4);
  float* t0    = (float*)carve((size_t)NR * MAXNS * 4);
  float* t1    = (float*)carve((size_t)NR * MAXNS * 4);
  float* s0    = (float*)carve((size_t)NR * MAXNS * 4);
  float* s1    = (float*)carve((size_t)NR * MAXNS * 4);
  int*   maskb = (int*)  carve((size_t)NR * MAXNS * 4);
  int*   eok   = (int*)  carve((size_t)NR * 4);
  int*   nbb   = (int*)  carve((size_t)NR * MAXNS * 4);
  float* frb   = (float*)carve((size_t)NR * MAXNS * 4);
  float* ptsb  = (float*)carve((size_t)NR * MAXNS * 3 * 4);
  __bf16* W1t  = (__bf16*)carve((size_t)HID * HID * 2);
  __bf16* W2t  = (__bf16*)carve((size_t)HID * HID * 2);

  auto run_iter = [&](float* t_cur, float* s_cur, int Ns, bool do_eok) {
    int npts = NR * Ns;
    k_gen_points<<<(npts + 255) / 256, 256, 0, stream>>>(cam, dirs, t_cur, ptsb, Ns);
    k_mlp<<<npts / MLP_PTS, MLP_TPB, 0, stream>>>(ptsb, maskb, s_cur,
                                                  W0, b0, W1t, b1, W2t, b2, W3, b3);
    if (do_eok)
      k_error_ok<<<(NR + 255) / 256, 256, 0, stream>>>(s_cur, t_cur, beta, Ns, eok);
  };

  k_convert<<<(2 * HID * HID + 255) / 256, 256, 0, stream>>>(W1, W2, W1t, W2t);
  k_init<<<(NR + 255) / 256, 256, 0, stream>>>(cam, ipts, insrc, dirs, t0, s0, maskb);

  run_iter(t0, s0, 64, true);                                             // i = 0
  k_upsample<<<(NR + 255) / 256, 256, 0, stream>>>(s0, t0, eok, insrc, beta, 64,
                                                   s1, t1, maskb, nbb, frb);
  run_iter(t1, s1, 128, true);                                            // i = 1
  k_upsample<<<(NR + 255) / 256, 256, 0, stream>>>(s1, t1, eok, insrc, beta, 128,
                                                   s0, t0, maskb, nbb, frb);
  run_iter(t0, s0, 192, true);                                            // i = 2
  k_upsample<<<(NR + 255) / 256, 256, 0, stream>>>(s0, t0, eok, insrc, beta, 192,
                                                   s1, t1, maskb, nbb, frb);
  run_iter(t1, s1, 256, false);                                           // i = 3

  k_final<<<(NR + 255) / 256, 256, 0, stream>>>(s1, t1, insrc, beta, 256, (float*)d_out);
}
